// DroneDetectorMamba_27315992003102
// MI455X (gfx1250) — matrix operand order repaired and checked by reading, compiled, not run on hardware
//
#include <hip/hip_runtime.h>
#include <math.h>

typedef unsigned short bf16_t;

// ---------------------------------------------------------------------------
// Types for CDNA5 WMMA
// ---------------------------------------------------------------------------
typedef __attribute__((ext_vector_type(16))) __bf16 v16bf;
typedef __attribute__((ext_vector_type(8)))  float  v8f;

union Frag16 {
    v16bf v;
    uint4 q[2];
    unsigned int u[8];
};

__device__ __forceinline__ unsigned short f2bf(float a) {
    unsigned int ua = __float_as_uint(a);
    return (unsigned short)((ua + 0x7fffu + ((ua >> 16) & 1u)) >> 16);
}

__device__ __forceinline__ float siluf(float x) {
    return x / (1.0f + __expf(-x));
}

// A-fragment: lane holds row m; VGPR v -> K = (v>=4?16:0) + half*8 + (v&3)*2
// bf16 memory spans: [k0+half*8, +8) and [k0+16+half*8, +8) -> 2 x b128
__device__ __forceinline__ Frag16 load_frag_a(const bf16_t* __restrict__ A,
                                              size_t row, int lda, int k0, int half) {
    const bf16_t* p = A + row * (size_t)lda + k0 + half * 8;
    Frag16 f;
    f.q[0] = *(const uint4*)(p);
    f.q[1] = *(const uint4*)(p + 16);
    return f;
}

// B-fragment: lane holds column n; VGPR v -> K = (v>=4?16:0) + half*8 + (v&3)*2
// bf16 memory spans: [k0+half*8, +8) and [k0+16+half*8, +8) -> 2 x b128
__device__ __forceinline__ Frag16 load_frag_b(const bf16_t* __restrict__ W,
                                              size_t col, int K, int k0, int half) {
    const bf16_t* p = W + col * (size_t)K + k0 + half * 8;
    Frag16 f;
    f.q[0] = *(const uint4*)(p);
    f.q[1] = *(const uint4*)(p + 16);
    return f;
}

// ---------------------------------------------------------------------------
// f32 -> bf16 (RNE), 2 elements per thread, dword stores. n must be even.
// ---------------------------------------------------------------------------
__global__ void f32_to_bf16(const float* __restrict__ in, bf16_t* __restrict__ out, int n) {
    int i = (blockIdx.x * blockDim.x + threadIdx.x) * 2;
    if (i >= n) return;
    unsigned int lo = f2bf(in[i]);
    unsigned int hi = f2bf(in[i + 1]);
    *(unsigned int*)(out + i) = lo | (hi << 16);
}

// ---------------------------------------------------------------------------
// Fused Conv3x3(SAME) + BatchNorm(eval) + ReLU + MaxPool2x2 (stride 2)
// ---------------------------------------------------------------------------
__global__ void conv_bn_relu_pool(const float* __restrict__ in,
                                  const float* __restrict__ w,
                                  const float* __restrict__ g,
                                  const float* __restrict__ b,
                                  float* __restrict__ out,
                                  int N, int Cin, int Cout, int H, int W) {
    const int OW = W >> 1, OH = H >> 1;
    int idx = blockIdx.x * blockDim.x + threadIdx.x;
    int total = N * Cout * OH * OW;
    if (idx >= total) return;
    int pw = idx % OW;
    int t  = idx / OW;
    int ph = t % OH;  t /= OH;
    int co = t % Cout;
    int n  = t / Cout;

    const float scale = g[co] * rsqrtf(1.0f + 1e-5f);
    const float bias  = b[co];

    float mx = 0.0f;  // ReLU outputs are >= 0
    for (int dy = 0; dy < 2; ++dy) {
        for (int dx = 0; dx < 2; ++dx) {
            const int oy = 2 * ph + dy;
            const int ox = 2 * pw + dx;
            float s = 0.0f;
            for (int ci = 0; ci < Cin; ++ci) {
                const float* ip = in + ((size_t)(n * Cin + ci) * H) * W;
                const float* wp = w + (size_t)(co * Cin + ci) * 9;
                #pragma unroll
                for (int ky = 0; ky < 3; ++ky) {
                    const int iy = oy + ky - 1;
                    if (iy < 0 || iy >= H) continue;
                    #pragma unroll
                    for (int kx = 0; kx < 3; ++kx) {
                        const int ix = ox + kx - 1;
                        if (ix < 0 || ix >= W) continue;
                        s += ip[(size_t)iy * W + ix] * wp[ky * 3 + kx];
                    }
                }
            }
            float v = fmaxf(s * scale + bias, 0.0f);
            mx = fmaxf(mx, v);
        }
    }
    out[idx] = mx;
}

// ---------------------------------------------------------------------------
// [B,256,8,64] -> tokens [B,64,2048] with d = c*8 + mel
// ---------------------------------------------------------------------------
__global__ void to_tokens(const float* __restrict__ p4, float* __restrict__ tok, int B) {
    int idx = blockIdx.x * blockDim.x + threadIdx.x;
    int total = B * 64 * 2048;
    if (idx >= total) return;
    int d = idx % 2048;
    int t = (idx / 2048) % 64;
    int b = idx / (2048 * 64);
    int c = d >> 3, m = d & 7;
    tok[idx] = p4[(((size_t)b * 256 + c) * 8 + m) * 64 + t];
}

// ---------------------------------------------------------------------------
// WMMA bf16 GEMM:  C[M,N] = act( A[M,K(lda)] * W[N,K]^T + bias[N] )
// A, W pre-converted to bf16. act: 0 none, 1 relu, 2 softplus.
// Wave computes 64(M) x 16(N): 4 accumulators share one B fragment per chunk.
// K % 32 == 0, lda % 8 == 0. Loads use clamped indices (no divergent branches);
// garbage rows/cols land only in outputs we never store.
// ---------------------------------------------------------------------------
__global__ void gemm_bf16_wmma(const bf16_t* __restrict__ A,
                               const bf16_t* __restrict__ W,
                               const float* __restrict__ bias,
                               float* __restrict__ C,
                               int M, int N, int K, int lda, int ldc, int act) {
    const int waveId = (blockIdx.x * blockDim.x + threadIdx.x) >> 5;
    const int lane   = threadIdx.x & 31;
    const int nTiles  = (N + 15) >> 4;
    const int mGroups = (M + 63) >> 6;
    if (waveId >= nTiles * mGroups) return;   // wave-uniform: EXEC stays all-ones
    const int nt = waveId % nTiles;
    const int mg = waveId / nTiles;

    const int r    = lane & 15;
    const int half = lane >> 4;

    const size_t nL  = (size_t)min(nt * 16 + r, N - 1);
    const size_t m0L = (size_t)min(mg * 64 + r,      M - 1);
    const size_t m1L = (size_t)min(mg * 64 + 16 + r, M - 1);
    const size_t m2L = (size_t)min(mg * 64 + 32 + r, M - 1);
    const size_t m3L = (size_t)min(mg * 64 + 48 + r, M - 1);

    v8f acc0 = {}, acc1 = {}, acc2 = {}, acc3 = {};

    for (int k0 = 0; k0 < K; k0 += 32) {
        Frag16 bf = load_frag_b(W, nL, K, k0, half);
        Frag16 a0 = load_frag_a(A, m0L, lda, k0, half);
        Frag16 a1 = load_frag_a(A, m1L, lda, k0, half);
        Frag16 a2 = load_frag_a(A, m2L, lda, k0, half);
        Frag16 a3 = load_frag_a(A, m3L, lda, k0, half);
        acc0 = __builtin_amdgcn_wmma_f32_16x16x32_bf16(false, a0.v, false, bf.v, (short)0, acc0, false, false);
        acc1 = __builtin_amdgcn_wmma_f32_16x16x32_bf16(false, a1.v, false, bf.v, (short)0, acc1, false, false);
        acc2 = __builtin_amdgcn_wmma_f32_16x16x32_bf16(false, a2.v, false, bf.v, (short)0, acc2, false, false);
        acc3 = __builtin_amdgcn_wmma_f32_16x16x32_bf16(false, a3.v, false, bf.v, (short)0, acc3, false, false);
    }

    // ---- epilogue: C/D layout: VGPR v -> M = v + half*8, N = lane%16
    const int n = nt * 16 + r;
    float bv = 0.0f;
    if (bias != nullptr && n < N) bv = bias[n];

    v8f accs[4] = {acc0, acc1, acc2, acc3};
    #pragma unroll
    for (int t = 0; t < 4; ++t) {
        #pragma unroll
        for (int v = 0; v < 8; ++v) {
            const int m = mg * 64 + t * 16 + v + half * 8;
            if (m < M && n < N) {
                float val = accs[t][v] + bv;
                if (act == 1)      val = fmaxf(val, 0.0f);
                else if (act == 2) val = (val > 20.0f) ? val : log1pf(__expf(val));
                C[(size_t)m * ldc + n] = val;
            }
        }
    }
}

// ---------------------------------------------------------------------------
// Causal depthwise conv1d (width 4) + bias + SiLU.
// ---------------------------------------------------------------------------
__global__ void dwconv_silu(const float* __restrict__ xz,
                            const float* __restrict__ cw,
                            const float* __restrict__ cb,
                            float* __restrict__ xc,
                            int B, int L, int D) {
    int idx = blockIdx.x * blockDim.x + threadIdx.x;
    int total = B * L * D;
    if (idx >= total) return;
    int d = idx % D;
    int t = (idx / D) % L;
    int b = idx / (D * L);
    float s = cb[d];
    #pragma unroll
    for (int j = 0; j < 4; ++j) {
        int tt = t - 3 + j;
        if (tt >= 0)
            s += cw[d * 4 + j] * xz[((size_t)(b * L + tt)) * 8192 + d];
    }
    xc[idx] = siluf(s);
}

// ---------------------------------------------------------------------------
// Selective scan: one thread per (b,d); 16-state recurrence over L steps.
// xdbl rows stride 160: dt=[0:128), B=[128:144), C=[144:160)
// ---------------------------------------------------------------------------
__global__ void selective_scan(const float* __restrict__ xc,
                               const float* __restrict__ dt,
                               const float* __restrict__ xdbl,
                               const float* __restrict__ A_log,
                               const float* __restrict__ Dvec,
                               float* __restrict__ ys,
                               int B, int L, int D) {
    int idx = blockIdx.x * blockDim.x + threadIdx.x;
    if (idx >= B * D) return;
    int d = idx % D, b = idx / D;
    float A[16], h[16];
    #pragma unroll
    for (int nn = 0; nn < 16; ++nn) {
        A[nn] = -__expf(A_log[d * 16 + nn]);
        h[nn] = 0.0f;
    }
    const float Dd = Dvec[d];
    for (int l = 0; l < L; ++l) {
        const int tb = b * L + l;
        const float dtv = dt[(size_t)tb * D + d];
        const float xv  = xc[(size_t)tb * D + d];
        const float* Br = xdbl + (size_t)tb * 160 + 128;
        const float* Cr = Br + 16;
        float y = 0.0f;
        #pragma unroll
        for (int nn = 0; nn < 16; ++nn) {
            const float dA = __expf(dtv * A[nn]);
            h[nn] = dA * h[nn] + dtv * Br[nn] * xv;
            y += h[nn] * Cr[nn];
        }
        ys[(size_t)tb * D + d] = y + xv * Dd;
    }
}

// ---------------------------------------------------------------------------
__global__ void gate_silu(float* __restrict__ ys, const float* __restrict__ xz,
                          int BL, int D) {
    int idx = blockIdx.x * blockDim.x + threadIdx.x;
    if (idx >= BL * D) return;
    int d = idx % D;
    int t = idx / D;
    float z = xz[(size_t)t * 8192 + 4096 + d];
    ys[idx] *= siluf(z);
}

// ---------------------------------------------------------------------------
__global__ void mean_t(const float* __restrict__ in, float* __restrict__ out,
                       int B, int L, int Dm) {
    int idx = blockIdx.x * blockDim.x + threadIdx.x;
    if (idx >= B * Dm) return;
    int d = idx % Dm, b = idx / Dm;
    float s = 0.0f;
    for (int l = 0; l < L; ++l)
        s += in[((size_t)(b * L + l)) * Dm + d];
    out[idx] = s / (float)L;
}

// ---------------------------------------------------------------------------
static inline int ceil_div(int a, int b) { return (a + b - 1) / b; }

extern "C" void kernel_launch(void* const* d_in, const int* in_sizes, int n_in,
                              void* d_out, int out_size, void* d_ws, size_t ws_size,
                              hipStream_t stream) {
    (void)in_sizes; (void)n_in; (void)out_size; (void)ws_size;

    // ---- inputs (setup_inputs dict insertion order, depth-first) ----
    const float* x  = (const float*)d_in[0];
    const float* cW[4] = {(const float*)d_in[1], (const float*)d_in[4],
                          (const float*)d_in[7], (const float*)d_in[10]};
    const float* bnG[4] = {(const float*)d_in[2], (const float*)d_in[5],
                           (const float*)d_in[8], (const float*)d_in[11]};
    const float* bnB[4] = {(const float*)d_in[3], (const float*)d_in[6],
                           (const float*)d_in[9], (const float*)d_in[12]};
    const int M1 = 13, M2 = 22;  // mamba param bases
    const float* cls_w1 = (const float*)d_in[31];
    const float* cls_b1 = (const float*)d_in[32];
    const float* cls_w2 = (const float*)d_in[33];
    const float* cls_b2 = (const float*)d_in[34];
    const float* cls_w3 = (const float*)d_in[35];
    const float* cls_b3 = (const float*)d_in[36];

    // ---- workspace layout: f32 region (16.78M floats) + bf16 slots ----
    float* ws = (float*)d_ws;
    float* p1   = ws + 0;           // 8,388,608  [8,32,64,512]
    float* p2   = ws + 8388608;     // 4,194,304  [8,64,32,256]
    float* p3   = ws + 12582912;    // 2,097,152  [8,128,16,128]
    float* p4   = ws + 14680064;    // 1,048,576  [8,256,8,64]
    float* tokE = ws + 15728640;    // 1,048,576  tokens; later mamba2 output
    // mamba scratch (reuses p1/p2 regions after convs are consumed)
    float* xzB  = ws + 0;           // 4,194,304  [512,8192]
    float* xcB  = ws + 4194304;     // 2,097,152  [512,4096]
    float* dtB  = ws + 6291456;     // 2,097,152  [512,4096]
    float* ysB  = ws + 8388608;     // 2,097,152  [512,4096]
    float* xdB  = ws + 10485760;    //    81,920  [512,160]
    float* mo1  = ws + 10567680;    // 1,048,576  mamba1 output [512,2048]
    // classifier scratch (reuses p3 region)
    float* pooled = ws + 12582912;  // 16,384 [8,2048]
    float* h1     = ws + 12599296;  // 1,024  [8,128]
    float* h2     = ws + 12600320;  // 512    [8,64]
    // bf16 slots (shared across sequential GEMMs)
    bf16_t* wbf = (bf16_t*)(ws + 16777216);   // 16,777,216 elems (largest weight)
    bf16_t* abf = wbf + 16777216;             //  2,097,152 elems (largest activation)

    auto cvt = [&](const float* src, bf16_t* dst, int n) {
        f32_to_bf16<<<ceil_div(n / 2, 256), 256, 0, stream>>>(src, dst, n);
    };

    // A [M x lda] f32, W [N x K] f32 -> bf16 slots, then WMMA GEMM.
    auto gemm = [&](const float* A, const float* Wm, const float* bias, float* C,
                    int M, int N, int K, int lda, int ldc, int act) {
        cvt(A, abf, M * lda);
        cvt(Wm, wbf, N * K);
        int waves  = ceil_div(M, 64) * ceil_div(N, 16);
        int blocks = ceil_div(waves, 8);  // 8 waves (256 threads) per block
        gemm_bf16_wmma<<<blocks, 256, 0, stream>>>(abf, wbf, bias, C, M, N, K, lda, ldc, act);
    };

    // ---- conv frontend (fused conv+BN+ReLU+pool) ----
    conv_bn_relu_pool<<<ceil_div(8 * 32 * 64 * 512, 256), 256, 0, stream>>>(
        x, cW[0], bnG[0], bnB[0], p1, 8, 1, 32, 128, 1024);
    conv_bn_relu_pool<<<ceil_div(8 * 64 * 32 * 256, 256), 256, 0, stream>>>(
        p1, cW[1], bnG[1], bnB[1], p2, 8, 32, 64, 64, 512);
    conv_bn_relu_pool<<<ceil_div(8 * 128 * 16 * 128, 256), 256, 0, stream>>>(
        p2, cW[2], bnG[2], bnB[2], p3, 8, 64, 128, 32, 256);
    conv_bn_relu_pool<<<ceil_div(8 * 256 * 8 * 64, 256), 256, 0, stream>>>(
        p3, cW[3], bnG[3], bnB[3], p4, 8, 128, 256, 16, 128);
    to_tokens<<<ceil_div(8 * 64 * 2048, 256), 256, 0, stream>>>(p4, tokE, 8);

    // ---- mamba block ----
    auto mamba = [&](const float* X, float* outp, int base) {
        const float* in_proj_w  = (const float*)d_in[base + 0];
        const float* conv_w     = (const float*)d_in[base + 1];
        const float* conv_b     = (const float*)d_in[base + 2];
        const float* x_proj_w   = (const float*)d_in[base + 3];
        const float* dt_proj_w  = (const float*)d_in[base + 4];
        const float* dt_proj_b  = (const float*)d_in[base + 5];
        const float* A_log      = (const float*)d_in[base + 6];
        const float* Dvec       = (const float*)d_in[base + 7];
        const float* out_proj_w = (const float*)d_in[base + 8];
        const int BL = 512;  // 8 * 64 tokens

        gemm(X, in_proj_w, nullptr, xzB, BL, 8192, 2048, 2048, 8192, 0);
        dwconv_silu<<<ceil_div(BL * 4096, 256), 256, 0, stream>>>(
            xzB, conv_w, conv_b, xcB, 8, 64, 4096);
        gemm(xcB, x_proj_w, nullptr, xdB, BL, 160, 4096, 4096, 160, 0);
        gemm(xdB, dt_proj_w, dt_proj_b, dtB, BL, 4096, 128, 160, 4096, /*softplus*/2);
        selective_scan<<<ceil_div(8 * 4096, 256), 256, 0, stream>>>(
            xcB, dtB, xdB, A_log, Dvec, ysB, 8, 64, 4096);
        gate_silu<<<ceil_div(BL * 4096, 256), 256, 0, stream>>>(ysB, xzB, BL, 4096);
        gemm(ysB, out_proj_w, nullptr, outp, BL, 2048, 4096, 4096, 2048, 0);
    };

    mamba(tokE, mo1, M1);   // block 1: tokens -> mo1
    mamba(mo1, tokE, M2);   // block 2: mo1 -> tokE

    // ---- head ----
    mean_t<<<ceil_div(8 * 2048, 256), 256, 0, stream>>>(tokE, pooled, 8, 64, 2048);
    gemm(pooled, cls_w1, cls_b1, h1, 8, 128, 2048, 2048, 128, /*relu*/1);
    gemm(h1, cls_w2, cls_b2, h2, 8, 64, 128, 128, 64, /*relu*/1);
    gemm(h2, cls_w3, cls_b3, (float*)d_out, 8, 2, 64, 64, 2, 0);
}
